// VectorQuantizer_35966056136995
// MI455X (gfx1250) — matrix operand order repaired and checked by reading, not compiled, not run on hardware
//
#include <hip/hip_runtime.h>

#define NUM_EMB 1024
#define EMB_DIM 256
#define NPIX    32768                    // 32*32*32 vectors
#define OUT_ELEMS (NPIX * EMB_DIM)       // 8388608
#define ROWS_PER_BLOCK 32                // 2 M-subtiles of 16
#define NBLOCKS (NPIX / ROWS_PER_BLOCK)  // 1024
#define NTILES  (NUM_EMB / 16)           // 64 column tiles
#define KSTEPS  (EMB_DIM / 32)           // 8 k-steps of 32
#define DPITCH  132                      // 128 + 4 pad score slab
#define AFRAG_DWORDS (2 * KSTEPS * 32 * 8)   // 4096 dwords = 16KB per hi/lo

typedef _Float16 v16h __attribute__((ext_vector_type(16)));
typedef float    v8f  __attribute__((ext_vector_type(8)));

__device__ __forceinline__ unsigned pack2h(_Float16 a, _Float16 b) {
    unsigned ua = __builtin_bit_cast(unsigned short, a);
    unsigned ub = __builtin_bit_cast(unsigned short, b);
    return ua | (ub << 16);
}

__device__ __forceinline__ v8f wmma16(v16h a, v16h b, v8f c) {
    return __builtin_amdgcn_wmma_f32_16x16x32_f16(
        false, a, false, b, (short)0, c, false, false);
}

// ---------------------------------------------------------------------------
// Kernel 1a: codebook column squared norms (exact f32)
// ---------------------------------------------------------------------------
__global__ __launch_bounds__(256) void vq_enorm_kernel(const float* __restrict__ E,
                                                       float* __restrict__ enorm) {
    int j = blockIdx.x * 256 + threadIdx.x;
    float s = 0.0f;
#pragma unroll 8
    for (int k = 0; k < EMB_DIM; ++k) {
        float e = E[k * NUM_EMB + j];
        s += e * e;
    }
    enorm[j] = s;
}

// ---------------------------------------------------------------------------
// Kernel 1b: pre-swizzle E into hi/lo split-f16 B fragments.
// B layout (16-bit, 32x16, wave32): lane = h*16 + n15 (h=K-half), N = nt*16+n15,
// VGPR/dword d holds K = ks*32 + 2d + 16h  and  K+1 (packed lo16/hi16).
// Fragment (nt,ks) for lane L lives at dwords [((nt*8+ks)*32+L)*8 .. +7].
// ---------------------------------------------------------------------------
__global__ __launch_bounds__(256) void vq_eprep_kernel(const float* __restrict__ E,
                                                       unsigned* __restrict__ Ehi,
                                                       unsigned* __restrict__ Elo) {
    int id = blockIdx.x * 256 + threadIdx.x;   // 0 .. 131071
    int d    = id & 7;
    int lane = (id >> 3) & 31;
    int ks   = (id >> 8) & 7;
    int nt   = id >> 11;
    int h    = lane >> 4;
    int n    = nt * 16 + (lane & 15);
    int k0   = ks * 32 + 2 * d + 16 * h;
    float v0 = E[k0 * NUM_EMB + n];
    float v1 = E[(k0 + 1) * NUM_EMB + n];
    _Float16 h0 = (_Float16)v0, h1 = (_Float16)v1;
    _Float16 l0 = (_Float16)(v0 - (float)h0);
    _Float16 l1 = (_Float16)(v1 - (float)h1);
    Ehi[id] = pack2h(h0, h1);
    Elo[id] = pack2h(l0, l1);
}

// ---------------------------------------------------------------------------
// Kernel 2: 32 rows/block; split-f16 WMMA scores; argmin; gather; loss partial
// ---------------------------------------------------------------------------
__global__ __launch_bounds__(256) void vq_main_kernel(const float* __restrict__ X,
                                                      const float* __restrict__ E,
                                                      const unsigned* __restrict__ EhiSwz,
                                                      const unsigned* __restrict__ EloSwz,
                                                      const float* __restrict__ enorm,
                                                      float* __restrict__ out,
                                                      float* __restrict__ partial) {
    __shared__ unsigned Ahi[AFRAG_DWORDS];       // 16 KB: A fragments, hi half
    __shared__ unsigned Alo[AFRAG_DWORDS];       // 16 KB: A fragments, lo half
    __shared__ float Ds[ROWS_PER_BLOCK * DPITCH];// ~16.9 KB score slab
    __shared__ float Rv[256];
    __shared__ int   Ri[256];
    __shared__ int   rowIdx[ROWS_PER_BLOCK];
    __shared__ float red[256];

    const int tid     = threadIdx.x;
    const int lane    = tid & 31;     // wave32
    const int wave    = tid >> 5;     // 8 waves
    const int rowbase = blockIdx.x * ROWS_PER_BLOCK;

    // Stage X tile as split-f16 A fragments (16x32 A layout, contiguous k order matching B):
    // lane = h*16 + m15, M = mt*16 + m15,
    // dword d: K = ks*32 + 2d + 16h, packed K/K+1.
    for (int i = tid; i < AFRAG_DWORDS; i += 256) {
        int d  = i & 7;
        int ln = (i >> 3) & 31;
        int ks = (i >> 8) & 7;
        int mt = i >> 11;
        int h  = ln >> 4;
        int m  = mt * 16 + (ln & 15);
        int k0 = ks * 32 + 2 * d + 16 * h;
        float v0 = X[(rowbase + m) * EMB_DIM + k0];
        float v1 = X[(rowbase + m) * EMB_DIM + k0 + 1];
        _Float16 h0 = (_Float16)v0, h1 = (_Float16)v1;
        Ahi[i] = pack2h(h0, h1);
        Alo[i] = pack2h((_Float16)(v0 - (float)h0), (_Float16)(v1 - (float)h1));
    }
    __syncthreads();

    const v16h* Bh = (const v16h*)EhiSwz;
    const v16h* Bl = (const v16h*)EloSwz;
    const v16h* Ah = (const v16h*)Ahi;
    const v16h* Al = (const v16h*)Alo;

    const int l15 = lane & 15;
    const int hh  = lane >> 4;

    // running argmin: thread t owns row t>>3, 16-col slice (t&7) of each chunk
    float bestVal = 3.402823466e38f;
    int   bestIdx = 0;
    const int myrow = tid >> 3;
    const int myj   = tid & 7;

    for (int c = 0; c < 8; ++c) {              // 8 chunks x 128 columns
        const int nt = c * 8 + wave;           // this wave's 16-col tile
        v8f acc0 = {}, acc1 = {};
#pragma unroll
        for (int ks = 0; ks < KSTEPS; ++ks) {
            const int bf = (nt * KSTEPS + ks) * 32 + lane;
            v16h bh  = Bh[bf];
            v16h bl  = Bl[bf];
            const int a0 = (0 * KSTEPS + ks) * 32 + lane;
            const int a1 = (1 * KSTEPS + ks) * 32 + lane;
            v16h ah0 = Ah[a0];
            v16h al0 = Al[a0];
            v16h ah1 = Ah[a1];
            v16h al1 = Al[a1];
            // Markidis split: X.E = Xh.Eh + Xh.El + Xl.Eh  (f32 accumulate)
            acc0 = wmma16(ah0, bh, acc0);
            acc0 = wmma16(ah0, bl, acc0);
            acc0 = wmma16(al0, bh, acc0);
            acc1 = wmma16(ah1, bh, acc1);
            acc1 = wmma16(ah1, bl, acc1);
            acc1 = wmma16(al1, bh, acc1);
        }

        // scores = ||e||^2 - 2*(x.e)  ->  slab (C/D layout: vgpr r, half hh)
        const float en = enorm[nt * 16 + l15];
#pragma unroll
        for (int r = 0; r < 8; ++r) {
            const int colb = wave * 16 + l15;
            Ds[(r + 8 * hh) * DPITCH + colb]      = en - 2.0f * acc0[r];
            Ds[(16 + r + 8 * hh) * DPITCH + colb] = en - 2.0f * acc1[r];
        }
        __syncthreads();

        // update running argmin (strict < keeps earliest index, like argmin)
#pragma unroll
        for (int i = 0; i < 16; ++i) {
            const int lc = myj * 16 + i;
            const float v = Ds[myrow * DPITCH + lc];
            if (v < bestVal) { bestVal = v; bestIdx = c * 128 + lc; }
        }
        __syncthreads();
    }

    // combine the 8 column-slices per row (tie-break: lowest index)
    Rv[tid] = bestVal;
    Ri[tid] = bestIdx;
    __syncthreads();
    if (tid < ROWS_PER_BLOCK) {
        float bv = Rv[tid * 8];
        int   bi = Ri[tid * 8];
#pragma unroll
        for (int j = 1; j < 8; ++j) {
            float v = Rv[tid * 8 + j];
            int   ii = Ri[tid * 8 + j];
            if (v < bv || (v == bv && ii < bi)) { bv = v; bi = ii; }
        }
        rowIdx[tid] = bi;
    }
    __syncthreads();

    // gather winning codebook columns (exact f32 E), output, loss partial
    float sq = 0.0f;
#pragma unroll 2
    for (int r = 0; r < ROWS_PER_BLOCK; ++r) {
        const int   idx = rowIdx[r];
        const float q   = E[tid * NUM_EMB + idx];
        const float x   = X[(rowbase + r) * EMB_DIM + tid];
        out[(rowbase + r) * EMB_DIM + tid] = x + (q - x);   // reference arith
        const float dd = x - q;
        sq += dd * dd;
    }
    red[tid] = sq;
    __syncthreads();
#pragma unroll
    for (int s = 128; s > 0; s >>= 1) {
        if (tid < s) red[tid] += red[tid + s];
        __syncthreads();
    }
    if (tid == 0) partial[blockIdx.x] = red[0];
}

// ---------------------------------------------------------------------------
// Kernel 3: deterministic loss reduction; loss = 1.25 * mean((x-q)^2)
// ---------------------------------------------------------------------------
__global__ __launch_bounds__(256) void vq_loss_kernel(const float* __restrict__ partial,
                                                      float* __restrict__ out) {
    __shared__ float red[256];
    float s = 0.0f;
    for (int i = threadIdx.x; i < NBLOCKS; i += 256) s += partial[i];
    red[threadIdx.x] = s;
    __syncthreads();
#pragma unroll
    for (int t = 128; t > 0; t >>= 1) {
        if (threadIdx.x < t) red[threadIdx.x] += red[threadIdx.x + t];
        __syncthreads();
    }
    if (threadIdx.x == 0)
        out[OUT_ELEMS] = 1.25f * red[0] * (1.0f / (float)OUT_ELEMS);
}

// ---------------------------------------------------------------------------
extern "C" void kernel_launch(void* const* d_in, const int* in_sizes, int n_in,
                              void* d_out, int out_size, void* d_ws, size_t ws_size,
                              hipStream_t stream) {
    const float* X = (const float*)d_in[0];   // (32,32,32,256) f32
    const float* E = (const float*)d_in[1];   // (256,1024) f32
    float* out = (float*)d_out;               // 8388608 out + 1 loss

    // workspace layout (all rewritten every launch; 0xAA poison is harmless)
    float*    enorm   = (float*)d_ws;                         // 4 KB
    unsigned* EhiSwz  = (unsigned*)(enorm + NUM_EMB);         // 512 KB
    unsigned* EloSwz  = EhiSwz + (NTILES * KSTEPS * 32 * 8);  // 512 KB
    float*    partial = (float*)(EloSwz + (NTILES * KSTEPS * 32 * 8)); // 4 KB

    vq_enorm_kernel<<<NUM_EMB / 256, 256, 0, stream>>>(E, enorm);
    vq_eprep_kernel<<<(NTILES * KSTEPS * 32 * 8) / 256, 256, 0, stream>>>(E, EhiSwz, EloSwz);
    vq_main_kernel<<<NBLOCKS, 256, 0, stream>>>(X, E, EhiSwz, EloSwz, enorm, out, partial);
    vq_loss_kernel<<<1, 256, 0, stream>>>(partial, out);
}
